// Attention_27367531611013
// MI455X (gfx1250) — compile-verified, matrix operand order checked by reading, not run on hardware
//
#include <hip/hip_runtime.h>
#include <hip/hip_bf16.h>

// ---------------------------------------------------------------------------
// Problem constants (from reference): B=128, S=1024, D=512, C=10
// Outputs: e [B, D, S] then logits [B, S], concatenated flat in d_out (f32).
// Workspace: qp [B*D] floats, u [B*S] floats  (768 KB total).
// ---------------------------------------------------------------------------

#define BB   128
#define SS   1024
#define DD   512
#define CLIP 10.0f

typedef __attribute__((ext_vector_type(16))) __bf16 v16bf;
typedef __attribute__((ext_vector_type(8)))  __bf16 v8bf;
typedef __attribute__((ext_vector_type(4)))  __bf16 v4bf;
typedef __attribute__((ext_vector_type(8)))  float  v8f;

// LDS tile row stride: 64 K-halfs + 8 pad = 72 halfs (144 B).
// 144 % 16 == 0 (b128-aligned); bank stride 36 dwords -> the 16 fragment rows
// hit 16 distinct banks (r*36 mod 64 is a permutation over r=0..15).
#define LDA 72

// ---------------------------------------------------------------------------
// Hardware tanh (CDNA5 TRANS op v_tanh_f32) with safe fallback.
// ---------------------------------------------------------------------------
__device__ __forceinline__ float fast_tanhf(float x) {
#if __has_builtin(__builtin_amdgcn_tanhf)
    return __builtin_amdgcn_tanhf(x);
#elif __has_builtin(__builtin_amdgcn_tanh_f32)
    return __builtin_amdgcn_tanh_f32(x);
#else
    return tanhf(x);
#endif
}

// ---------------------------------------------------------------------------
// Kernel 1: qp[b,o] = dot(q[b,:], Wq[o,:]) + bq[o]      (f32, exact)
// ---------------------------------------------------------------------------
__global__ void qp_proj_kernel(const float* __restrict__ q,
                               const float* __restrict__ Wq,
                               const float* __restrict__ bq,
                               float* __restrict__ qp) {
    int idx = blockIdx.x * 256 + threadIdx.x;   // 65536 outputs
    int b = idx >> 9;
    int o = idx & (DD - 1);
    const float4* qr = (const float4*)(q  + (size_t)b * DD);
    const float4* wr = (const float4*)(Wq + (size_t)o * DD);
    float s = 0.0f;
#pragma unroll 8
    for (int k = 0; k < DD / 4; ++k) {
        float4 a = qr[k], w = wr[k];
        s += a.x * w.x + a.y * w.y + a.z * w.z + a.w * w.w;
    }
    qp[idx] = s + bq[o];
}

// ---------------------------------------------------------------------------
// Kernel 2: zero the u accumulator (atomics accumulate into it each call)
// ---------------------------------------------------------------------------
__global__ void zero_u_kernel(float* __restrict__ u) {
    int idx = blockIdx.x * 256 + threadIdx.x;   // 131072
    u[idx] = 0.0f;
}

// ---------------------------------------------------------------------------
// LDS fragment fetch: two 16-byte ds loads per ISA 16-bit A/B layout.
// lane L holds row (L&15); K-halves (L>>4)*8 .. +7 and +16 .. +23.
// ---------------------------------------------------------------------------
__device__ __forceinline__ v16bf ldsFrag(const __bf16* base, int row, int kofs) {
    union { v16bf v; v8bf h[2]; } f;
    f.h[0] = *(const v8bf*)(base + row * LDA + kofs);
    f.h[1] = *(const v8bf*)(base + row * LDA + kofs + 16);
    return f.v;
}

__device__ __forceinline__ v4bf cvt4(float4 w) {
    v4bf r = { (__bf16)w.x, (__bf16)w.y, (__bf16)w.z, (__bf16)w.w };
    return r;
}

// ---------------------------------------------------------------------------
// Kernel 3: fused GEMM + bias + store e + v.tanh(qp+e) reduction into u.
//   grid = B * (S/64) * (D/128) blocks, 256 threads (8 waves).
//   Per block: 128(o) x 64(s) tile; double-buffered LDS, K=64 per iteration
//   (one barrier / 8 WMMAs per wave per iteration).
// ---------------------------------------------------------------------------
__global__ void __launch_bounds__(256)
attn_e_wmma_kernel(const float* __restrict__ ref,   // [S, B, D]
                   const float* __restrict__ Wr,    // [D, D]
                   const float* __restrict__ br,    // [D]
                   const float* __restrict__ vvec,  // [D]
                   const float* __restrict__ qp,    // [B, D]
                   float* __restrict__ u,           // [B, S] accumulator
                   float* __restrict__ eOut) {      // [B, D, S]
    __shared__ __bf16 aW[2][128 * LDA];   // Wr tiles  [128 o][64 k] bf16 (x2)
    __shared__ __bf16 bR[2][64  * LDA];   // ref tiles [ 64 s][64 k] bf16 (x2)

    const int tid  = threadIdx.x;
    const int lane = tid & 31;
    const int wave = tid >> 5;
    const int waveO = wave & 3;   // 4 waves along o (32 each)
    const int waveS = wave >> 2;  // 2 waves along s (32 each)

    const int bid   = blockIdx.x;
    const int ot    = bid & 3;           // D/128 = 4 o-tiles
    const int st    = (bid >> 2) & 15;   // S/64 = 16 s-tiles
    const int b     = bid >> 6;          // 128 batches
    const int oBase = ot * 128;
    const int sBase = st * 64;

    const int fragRow = lane & 15;
    const int kb      = (lane >> 4) * 8;

    // Cooperative-load geometry: 128x64 Wr tile = 2048 float4 (8/thread),
    //                            64x64 ref tile = 1024 float4 (4/thread).
    const int wRow = 0, wC4 = 0; (void)wRow; (void)wC4;

    const v8f vzero = { 0.f, 0.f, 0.f, 0.f, 0.f, 0.f, 0.f, 0.f };
    v8f acc[2][2];
    acc[0][0] = vzero; acc[0][1] = vzero; acc[1][0] = vzero; acc[1][1] = vzero;

    // ---- prologue: stage K-chunk 0 into buffer 0 --------------------------
    {
#pragma unroll
        for (int i = 0; i < 8; ++i) {
            int idx = tid + 256 * i;
            int row = idx >> 4, c4 = idx & 15;
            float4 w = *(const float4*)(Wr + (size_t)(oBase + row) * DD + c4 * 4);
            *(v4bf*)(&aW[0][row * LDA + c4 * 4]) = cvt4(w);
        }
#pragma unroll
        for (int i = 0; i < 4; ++i) {
            int idx = tid + 256 * i;
            int row = idx >> 4, c4 = idx & 15;
            float4 r4 = *(const float4*)(ref + (size_t)(sBase + row) * (BB * DD)
                                             + (size_t)b * DD + c4 * 4);
            *(v4bf*)(&bR[0][row * LDA + c4 * 4]) = cvt4(r4);
        }
    }
    __syncthreads();

    int cur = 0;
    for (int k0 = 0; k0 < DD; k0 += 64) {
        const int  nxt      = cur ^ 1;
        const bool has_next = (k0 + 64 < DD);

        // ---- issue global loads for the NEXT chunk into registers ---------
        float4 wr4[8], rf4[4];
        if (has_next) {
#pragma unroll
            for (int i = 0; i < 8; ++i) {
                int idx = tid + 256 * i;
                int row = idx >> 4, c4 = idx & 15;
                wr4[i] = *(const float4*)(Wr + (size_t)(oBase + row) * DD
                                             + k0 + 64 + c4 * 4);
            }
#pragma unroll
            for (int i = 0; i < 4; ++i) {
                int idx = tid + 256 * i;
                int row = idx >> 4, c4 = idx & 15;
                rf4[i] = *(const float4*)(ref + (size_t)(sBase + row) * (BB * DD)
                                              + (size_t)b * DD + k0 + 64 + c4 * 4);
            }
        }

        // ---- compute on the CURRENT buffer: 2 sub-chunks x 4 WMMAs --------
        const __bf16* aBuf = &aW[cur][0];
        const __bf16* bBuf = &bR[cur][0];
#pragma unroll
        for (int kk = 0; kk < 64; kk += 32) {
            v16bf af0 = ldsFrag(aBuf, waveO * 32 + fragRow,      kk + kb);
            v16bf af1 = ldsFrag(aBuf, waveO * 32 + 16 + fragRow, kk + kb);
            v16bf bf0 = ldsFrag(bBuf, waveS * 32 + fragRow,      kk + kb);
            v16bf bf1 = ldsFrag(bBuf, waveS * 32 + 16 + fragRow, kk + kb);

            acc[0][0] = __builtin_amdgcn_wmma_f32_16x16x32_bf16(false, af0, false, bf0,
                                                                (short)0, acc[0][0], false, false);
            acc[0][1] = __builtin_amdgcn_wmma_f32_16x16x32_bf16(false, af0, false, bf1,
                                                                (short)0, acc[0][1], false, false);
            acc[1][0] = __builtin_amdgcn_wmma_f32_16x16x32_bf16(false, af1, false, bf0,
                                                                (short)0, acc[1][0], false, false);
            acc[1][1] = __builtin_amdgcn_wmma_f32_16x16x32_bf16(false, af1, false, bf1,
                                                                (short)0, acc[1][1], false, false);
        }

        // ---- convert + park the next chunk in the OTHER buffer ------------
        if (has_next) {
#pragma unroll
            for (int i = 0; i < 8; ++i) {
                int idx = tid + 256 * i;
                int row = idx >> 4, c4 = idx & 15;
                *(v4bf*)(&aW[nxt][row * LDA + c4 * 4]) = cvt4(wr4[i]);
            }
#pragma unroll
            for (int i = 0; i < 4; ++i) {
                int idx = tid + 256 * i;
                int row = idx >> 4, c4 = idx & 15;
                *(v4bf*)(&bR[nxt][row * LDA + c4 * 4]) = cvt4(rf4[i]);
            }
        }
        __syncthreads();   // one barrier per K=64 iteration
        cur = nxt;
    }

    // ---- epilogue: bias, store e, fused v.tanh(qp+e) -> atomic u ----------
    const int hi   = lane >> 4;          // C/D layout: M = n + 8*hi, N = lane&15
    const int nCol = lane & 15;
#pragma unroll
    for (int j = 0; j < 2; ++j) {
        const int s = sBase + waveS * 32 + j * 16 + nCol;
        float tsum = 0.0f;
#pragma unroll
        for (int i = 0; i < 2; ++i) {
#pragma unroll
            for (int n = 0; n < 8; ++n) {
                const int o  = oBase + waveO * 32 + i * 16 + n + hi * 8;
                const float ev = acc[i][j][n] + br[o];
                eOut[((size_t)b * DD + o) * SS + s] = ev;
                tsum += vvec[o] * fast_tanhf(qp[b * DD + o] + ev);
            }
        }
        tsum += __shfl_xor(tsum, 16, 32);          // fold the two M-halves
        if (lane < 16)
            atomicAdd(&u[(size_t)b * SS + s], tsum);
    }
}

// ---------------------------------------------------------------------------
// Kernel 4: logits = C * tanh(u)
// ---------------------------------------------------------------------------
__global__ void logits_kernel(const float* __restrict__ u,
                              float* __restrict__ lg) {
    int idx = blockIdx.x * 256 + threadIdx.x;   // 131072
    lg[idx] = CLIP * fast_tanhf(u[idx]);
}

// ---------------------------------------------------------------------------
extern "C" void kernel_launch(void* const* d_in, const int* in_sizes, int n_in,
                              void* d_out, int out_size, void* d_ws, size_t ws_size,
                              hipStream_t stream) {
    const float* q   = (const float*)d_in[0];
    const float* ref = (const float*)d_in[1];
    const float* Wq  = (const float*)d_in[2];
    const float* bq  = (const float*)d_in[3];
    const float* Wr  = (const float*)d_in[4];
    const float* br  = (const float*)d_in[5];
    const float* v   = (const float*)d_in[6];

    float* eOut   = (float*)d_out;                              // B*D*S
    float* logits = (float*)d_out + (size_t)BB * DD * SS;       // B*S

    float* qp = (float*)d_ws;                                   // B*D floats
    float* u  = qp + (size_t)BB * DD;                           // B*S floats

    // 1) qp = q @ Wq^T + bq
    qp_proj_kernel<<<(BB * DD) / 256, 256, 0, stream>>>(q, Wq, bq, qp);
    // 2) zero u accumulator (fresh every call -> deterministic)
    zero_u_kernel<<<(BB * SS) / 256, 256, 0, stream>>>(u);
    // 3) fused e-GEMM (bf16 WMMA, double-buffered LDS) + bias + tanh reduce
    attn_e_wmma_kernel<<<BB * (SS / 64) * (DD / 128), 256, 0, stream>>>(
        ref, Wr, br, v, qp, u, eOut);
    // 4) logits = 10 * tanh(u)
    logits_kernel<<<(BB * SS) / 256, 256, 0, stream>>>(u, logits);
}
